// RiskPropagationGNN_77446850281569
// MI455X (gfx1250) — compile-verified, matrix operand order checked by reading, not run on hardware
//
#include <hip/hip_runtime.h>
#include <hip/hip_bf16.h>

// ---------------------------------------------------------------------------
// Types for CDNA5 WMMA (wave32): v16bf A/B operands, v8f accumulator
// ---------------------------------------------------------------------------
typedef __attribute__((ext_vector_type(16))) __bf16 v16bf;
typedef __attribute__((ext_vector_type(8)))  float  v8f;

// ---------------------------------------------------------------------------
// Helpers
// ---------------------------------------------------------------------------
__device__ __forceinline__ unsigned short f32_to_bf16_rne(float f) {
    unsigned u = __float_as_uint(f);
    u += 0x7FFFu + ((u >> 16) & 1u);   // round-to-nearest-even
    return (unsigned short)(u >> 16);
}

// order-preserving float -> uint mapping for atomicMax-based segment max
__device__ __forceinline__ unsigned enc_f32(float f) {
    unsigned u = __float_as_uint(f);
    return (u & 0x80000000u) ? ~u : (u | 0x80000000u);
}
__device__ __forceinline__ float dec_f32(unsigned k) {
    unsigned u = (k & 0x80000000u) ? (k & 0x7FFFFFFFu) : ~k;
    return __uint_as_float(u);
}

// ---------------------------------------------------------------------------
// Elementwise kernels
// ---------------------------------------------------------------------------
__global__ void k_cvt_bf16(const float* __restrict__ src,
                           unsigned short* __restrict__ dst, int n) {
    int i = blockIdx.x * blockDim.x + threadIdx.x;
    if (i < n) dst[i] = f32_to_bf16_rne(src[i]);
}

__global__ void k_zero_u32(unsigned* __restrict__ p, int n) {
    int i = blockIdx.x * blockDim.x + threadIdx.x;
    if (i < n) p[i] = 0u;
}

__global__ void k_bias_act(float* __restrict__ out, const float* __restrict__ b,
                           int n, int F, int do_relu) {
    int i = blockIdx.x * blockDim.x + threadIdx.x;
    if (i < n) {
        float v = out[i] + b[i % F];
        if (do_relu) v = v > 0.f ? v : 0.f;
        out[i] = v;
    }
}

// ---------------------------------------------------------------------------
// WMMA bf16 GEMM: C[M,N] = A[M,K] * B[K,N]
// A row-major bf16 (as ushort), B row-major bf16, C f32.
// Block: 256 threads = 8 waves; block tile 64(M) x 128(N); K-step 32.
// Wave grid 4(M) x 2(N); each wave: 16x64 strip = 4 x v_wmma_f32_16x16x32_bf16.
//
// LDS layouts (dwords, stride 20 = 16 used + 4 pad -> conflict-free b128 reads):
//   sA[row][k2]   : row-major A tile, k2 = K/2 dword index (natural pairs)
//   sB[n][k2]     : B tile pre-packed at store time: dword = (B[2k2][n], B[2k2+1][n])
// 16-bit operand ISA layout => lane(lg,lm) fragment = two contiguous b128 loads:
//   dwords [4*lg .. 4*lg+3] and [8+4*lg .. 8+4*lg+3].
// Software pipelined: next K-tile global loads issued before consuming LDS.
// Requires: K % 32 == 0, N % 128 == 0. M guarded.
// ---------------------------------------------------------------------------
__global__ __launch_bounds__(256) void k_gemm_wmma_bf16(
    const unsigned short* __restrict__ A,
    const unsigned short* __restrict__ B,
    float* __restrict__ Cmat,
    int M, int N, int K)
{
    __shared__ __align__(16) unsigned sA[64 * 20];
    __shared__ __align__(16) unsigned sB[128 * 20];

    const int tid  = threadIdx.x;
    const int lane = tid & 31;
    const int wid  = tid >> 5;
    const int wm   = wid & 3;     // wave row (M)
    const int wn   = wid >> 2;    // wave col (N)
    const int lg   = lane >> 4;   // lane group
    const int lm   = lane & 15;

    const int m0 = blockIdx.x * 64;
    const int n0 = blockIdx.y * 128;

    v8f acc[4];
    #pragma unroll
    for (int i = 0; i < 4; ++i)
        #pragma unroll
        for (int r = 0; r < 8; ++r) acc[i][r] = 0.f;

    // global->LDS mappings
    const int a_r  = tid >> 2;          // 0..63 (A row)
    const int a_cd = (tid & 3) * 4;     // dword offset in A row: 0,4,8,12
    const int b_k2 = tid >> 4;          // 0..15 (B k-pair)
    const int b_n  = (tid & 15) * 8;    // 0..120 (B col segment)

    uint4 ra, rb0, rb1;
    auto g_load = [&](int k0) {
        const int gr = m0 + a_r;
        ra = make_uint4(0u, 0u, 0u, 0u);
        if (gr < M) ra = *(const uint4*)(A + (size_t)gr * K + k0 + a_cd * 2);
        rb0 = *(const uint4*)(B + (size_t)(k0 + 2 * b_k2)     * N + n0 + b_n);
        rb1 = *(const uint4*)(B + (size_t)(k0 + 2 * b_k2 + 1) * N + n0 + b_n);
    };

    g_load(0);

    for (int k0 = 0; k0 < K; k0 += 32) {
        __syncthreads();
        // stage A (b128 store) and pre-packed B (transpose+pack, 8 b32 stores)
        *(uint4*)(sA + a_r * 20 + a_cd) = ra;
        {
            union { uint4 v; unsigned short s[8]; } u0, u1;
            u0.v = rb0; u1.v = rb1;
            #pragma unroll
            for (int j = 0; j < 8; ++j)
                sB[(b_n + j) * 20 + b_k2] =
                    (unsigned)u0.s[j] | ((unsigned)u1.s[j] << 16);
        }
        __syncthreads();

        // prefetch next K-tile while we compute on LDS
        if (k0 + 32 < K) g_load(k0 + 32);

        // A fragment: two ds_load_b128
        union { v16bf v; uint4 q[2]; } fa;
        const unsigned* arow = sA + (wm * 16 + lm) * 20 + lg * 4;
        fa.q[0] = *(const uint4*)(arow);
        fa.q[1] = *(const uint4*)(arow + 8);

        // 4 x (B fragment via two ds_load_b128 + WMMA)
        #pragma unroll
        for (int i = 0; i < 4; ++i) {
            union { v16bf v; uint4 q[2]; } fb;
            const unsigned* bcol = sB + (wn * 64 + i * 16 + lm) * 20 + lg * 4;
            fb.q[0] = *(const uint4*)(bcol);
            fb.q[1] = *(const uint4*)(bcol + 8);
            acc[i] = __builtin_amdgcn_wmma_f32_16x16x32_bf16(
                false, fa.v, false, fb.v, (short)0, acc[i], false, false);
        }
    }

    // store: VGPR r -> row (wm*16 + lg*8 + r), col = wn*64 + i*16 + lm
    #pragma unroll
    for (int i = 0; i < 4; ++i) {
        const int col = n0 + wn * 64 + i * 16 + lm;
        #pragma unroll
        for (int r = 0; r < 8; ++r) {
            const int row = m0 + wm * 16 + lg * 8 + r;
            if (row < M) Cmat[(size_t)row * N + col] = acc[i][r];
        }
    }
}

// ---------------------------------------------------------------------------
// Attention logits: al_s[n,h] = sum_c h[n,h,c]*a_src[h,c]; same for al_d
// ---------------------------------------------------------------------------
__global__ void k_alpha(const float* __restrict__ h,
                        const float* __restrict__ a_src,
                        const float* __restrict__ a_dst,
                        float* __restrict__ als, float* __restrict__ ald,
                        int N, int H, int C) {
    int idx = blockIdx.x * blockDim.x + threadIdx.x;
    if (idx >= N * H) return;
    int n = idx / H, hh = idx % H;
    const float* hp = h + (size_t)n * H * C + (size_t)hh * C;
    const float* as = a_src + hh * C;
    const float* ad = a_dst + hh * C;
    float ss = 0.f, sd = 0.f;
    for (int c = 0; c < C; ++c) { float v = hp[c]; ss += v * as[c]; sd += v * ad[c]; }
    als[idx] = ss;
    ald[idx] = sd;
}

// ---------------------------------------------------------------------------
// Edge pass 1: e = leaky_relu(al_s[src]+al_d[dst]); segment-max via atomicMax
// Edges 0..E-1 from edge_index (int64), edges E..E+N-1 are self loops.
// ---------------------------------------------------------------------------
__global__ void k_edge_max(const long long* __restrict__ ei,
                           const float* __restrict__ als,
                           const float* __restrict__ ald,
                           unsigned* __restrict__ mEnc,
                           float* __restrict__ ebuf,
                           int E, int N, int H) {
    int idx = blockIdx.x * blockDim.x + threadIdx.x;
    int Etot = E + N;
    if (idx >= Etot * H) return;
    int e = idx / H, hh = idx % H;
    int s = (e < E) ? (int)ei[e]     : (e - E);
    int d = (e < E) ? (int)ei[E + e] : (e - E);
    float v = als[s * H + hh] + ald[d * H + hh];
    v = v > 0.f ? v : 0.2f * v;             // leaky_relu, slope 0.2
    ebuf[idx] = v;
    atomicMax(&mEnc[d * H + hh], enc_f32(v));
}

// Edge pass 2: ex = exp(e - m[dst]); denom[dst] += ex
__global__ void k_edge_exp(const long long* __restrict__ ei,
                           float* __restrict__ ebuf,
                           const unsigned* __restrict__ mEnc,
                           float* __restrict__ denom,
                           int E, int N, int H) {
    int idx = blockIdx.x * blockDim.x + threadIdx.x;
    int Etot = E + N;
    if (idx >= Etot * H) return;
    int e = idx / H, hh = idx % H;
    int d = (e < E) ? (int)ei[E + e] : (e - E);
    float m  = dec_f32(mEnc[d * H + hh]);
    float ex = __expf(ebuf[idx] - m);
    ebuf[idx] = ex;
    atomicAdd(&denom[d * H + hh], ex);
}

// Edge pass 3: out[dst, c] += h[src, c] * (ex / denom[dst, head(c)])
// One block per edge; threads stride over F = H*C channels.
__global__ void k_edge_scatter(const long long* __restrict__ ei,
                               const float* __restrict__ hfeat,
                               const float* __restrict__ ebuf,
                               const float* __restrict__ denom,
                               float* __restrict__ out,
                               int E, int N, int H, int C) {
    int e = blockIdx.x;
    int F = H * C;
    int s = (e < E) ? (int)ei[e]     : (e - E);
    int d = (e < E) ? (int)ei[E + e] : (e - E);
    const float* hs = hfeat + (size_t)s * F;
    float*       od = out   + (size_t)d * F;
    for (int c = threadIdx.x; c < F; c += blockDim.x) {
        int hh = c / C;
        float a = ebuf[(size_t)e * H + hh] / denom[d * H + hh];
        atomicAdd(&od[c], hs[c] * a);
    }
}

// ---------------------------------------------------------------------------
// Risk head: out[n] = sigmoid( relu(h3[n] @ Ws1 + bs1) @ Ws2 + bs2 )
// h3: [N,128], Ws1: [128,64], Ws2: [64,1]. One block (64 threads) per node.
// ---------------------------------------------------------------------------
__global__ void k_head(const float* __restrict__ h3,
                       const float* __restrict__ Ws1, const float* __restrict__ bs1,
                       const float* __restrict__ Ws2, const float* __restrict__ bs2,
                       float* __restrict__ out, int N) {
    __shared__ float s[64];
    int n = blockIdx.x;
    int j = threadIdx.x;                       // 0..63
    const float* hv = h3 + (size_t)n * 128;
    float acc = bs1[j];
    for (int k = 0; k < 128; ++k) acc += hv[k] * Ws1[k * 64 + j];
    s[j] = acc > 0.f ? acc : 0.f;
    __syncthreads();
    if (j == 0) {
        float a2 = bs2[0];
        for (int t = 0; t < 64; ++t) a2 += s[t] * Ws2[t];
        out[n] = 1.f / (1.f + __expf(-a2));
    }
}

// ---------------------------------------------------------------------------
// Launch
// ---------------------------------------------------------------------------
extern "C" void kernel_launch(void* const* d_in, const int* in_sizes, int n_in,
                              void* d_out, int out_size, void* d_ws, size_t ws_size,
                              hipStream_t stream) {
    const float*     x   = (const float*)d_in[0];
    const long long* ei  = (const long long*)d_in[1];
    const float* W1  = (const float*)d_in[2];
    const float* as1 = (const float*)d_in[3];
    const float* ad1 = (const float*)d_in[4];
    const float* b1  = (const float*)d_in[5];
    const float* W2  = (const float*)d_in[6];
    const float* as2 = (const float*)d_in[7];
    const float* ad2 = (const float*)d_in[8];
    const float* b2  = (const float*)d_in[9];
    const float* W3  = (const float*)d_in[10];
    const float* as3 = (const float*)d_in[11];
    const float* ad3 = (const float*)d_in[12];
    const float* b3  = (const float*)d_in[13];
    const float* Ws1 = (const float*)d_in[14];
    const float* bs1 = (const float*)d_in[15];
    const float* Ws2 = (const float*)d_in[16];
    const float* bs2 = (const float*)d_in[17];

    const int D = 768, HC = 512, Hh = 4;
    const int N = in_sizes[0] / D;
    const int E = in_sizes[1] / 2;
    const int Etot = E + N;

    // workspace arena
    char* ws = (char*)d_ws;
    size_t off = 0;
    auto arena = [&](size_t bytes) -> char* {
        char* p = ws + off;
        off = (off + bytes + 255) & ~(size_t)255;
        return p;
    };
    unsigned short* bf_in  = (unsigned short*)arena((size_t)N * D * 2);     // max Fin
    unsigned short* bf_w   = (unsigned short*)arena((size_t)D * HC * 2);    // max weight
    float*    h_feat   = (float*)arena((size_t)N * HC * 4);
    float*    out_feat = (float*)arena((size_t)N * HC * 4);
    float*    als      = (float*)arena((size_t)N * Hh * 4);
    float*    ald      = (float*)arena((size_t)N * Hh * 4);
    float*    ebuf     = (float*)arena((size_t)Etot * Hh * 4);
    unsigned* mEnc     = (unsigned*)arena((size_t)N * Hh * 4);
    float*    denom    = (float*)arena((size_t)N * Hh * 4);

    auto cdiv = [](int a, int b) { return (a + b - 1) / b; };

    auto gat_layer = [&](const float* in_f, int Fin, int Fout, int H,
                         const float* W, const float* a_s, const float* a_d,
                         const float* b, int do_relu, float* out_f) {
        const int C = Fout / H;
        k_cvt_bf16<<<cdiv(N * Fin, 256), 256, 0, stream>>>(in_f, bf_in, N * Fin);
        k_cvt_bf16<<<cdiv(Fin * Fout, 256), 256, 0, stream>>>(W, bf_w, Fin * Fout);
        k_gemm_wmma_bf16<<<dim3(cdiv(N, 64), Fout / 128), 256, 0, stream>>>(
            bf_in, bf_w, h_feat, N, Fout, Fin);
        k_alpha<<<cdiv(N * H, 256), 256, 0, stream>>>(h_feat, a_s, a_d, als, ald, N, H, C);
        k_zero_u32<<<cdiv(N * H, 256), 256, 0, stream>>>(mEnc, N * H);
        k_zero_u32<<<cdiv(N * H, 256), 256, 0, stream>>>((unsigned*)denom, N * H);
        k_zero_u32<<<cdiv(N * Fout, 256), 256, 0, stream>>>((unsigned*)out_f, N * Fout);
        k_edge_max<<<cdiv(Etot * H, 256), 256, 0, stream>>>(ei, als, ald, mEnc, ebuf, E, N, H);
        k_edge_exp<<<cdiv(Etot * H, 256), 256, 0, stream>>>(ei, ebuf, mEnc, denom, E, N, H);
        int bs = Fout < 256 ? Fout : 256;
        k_edge_scatter<<<Etot, bs, 0, stream>>>(ei, h_feat, ebuf, denom, out_f, E, N, H, C);
        k_bias_act<<<cdiv(N * Fout, 256), 256, 0, stream>>>(out_f, b, N * Fout, Fout, do_relu);
    };

    // layer 1: [N,768] -> [N,512], relu
    gat_layer(x, D, HC, Hh, W1, as1, ad1, b1, 1, out_feat);
    // layer 2: [N,512] -> [N,512], relu  (out_feat consumed by cvt before reuse)
    gat_layer(out_feat, HC, HC, Hh, W2, as2, ad2, b2, 1, out_feat);
    // layer 3: [N,512] -> [N,128], H=1, no relu (mean over 1 head == identity)
    gat_layer(out_feat, HC, 128, 1, W3, as3, ad3, b3, 0, out_feat);

    // risk head -> d_out [N,1]
    k_head<<<N, 64, 0, stream>>>(out_feat, Ws1, bs1, Ws2, bs2, (float*)d_out, N);
}
